// MoLE_68659347194421
// MI455X (gfx1250) — matrix operand order repaired and checked by reading, not compiled, not run on hardware
//
#include <hip/hip_runtime.h>
#include <cstddef>

// ---------------------------------------------------------------------------
// MoLE (mixture of local experts) for MI455X / gfx1250, wave32 + WMMA f16.
//
// Phase 1: router logits  g[m][b][e] = dot(x[m][b], Wr[m][:,e])   (memory bound)
// Phase 2: gate           softmax/top2/modality-softmax           (tiny)
// Phase 3: fused experts  conv3x3(32->16)+ReLU -> conv3x3(16->32)+sigmoid,
//          8 evaluations accumulated per output tile via v_wmma_f32_16x16x32_f16
// ---------------------------------------------------------------------------

#define C_IN  32
#define C_HID 16
#define IMG_H 480
#define IMG_W 640
#define NPIX  (C_IN * IMG_H * IMG_W)   // 9,830,400 per image

typedef _Float16 v16h __attribute__((ext_vector_type(16)));
typedef _Float16 v8h  __attribute__((ext_vector_type(8)));
typedef float    v8f  __attribute__((ext_vector_type(8)));

union F16x16 { v16h v; v8h h8[2]; _Float16 h[16]; };
union F16x8  { v8h  v; _Float16 h[8]; };

__device__ __forceinline__ v8h zero8h() {
    v8h z = {(_Float16)0, (_Float16)0, (_Float16)0, (_Float16)0,
             (_Float16)0, (_Float16)0, (_Float16)0, (_Float16)0};
    return z;
}
__device__ __forceinline__ v8f zero8f() {
    v8f z = {0.f, 0.f, 0.f, 0.f, 0.f, 0.f, 0.f, 0.f};
    return z;
}
// reflect-pad index (np.pad mode='reflect'), valid for |overhang| <= 2
__device__ __forceinline__ int reflectI(int v, int n) {
    v = (v < 0) ? -v : v;
    return (v >= n) ? (2 * n - 2 - v) : v;
}

// ---------------------------------------------------------------------------
// workspace layout (floats): [0..15] logits (m*8+b*4+e), [16..19] wmod (m*2+b),
//                            [20..27] sel as int (m*2+b)*2+k
// ---------------------------------------------------------------------------

__global__ void zero_kernel(float* wsf) {
    if (threadIdx.x < 32) wsf[threadIdx.x] = 0.f;
}

// ------------------------- Phase 1: router ---------------------------------
__global__ __launch_bounds__(256)
void router_kernel(const float* __restrict__ xr, const float* __restrict__ xi,
                   const float* __restrict__ WrR, const float* __restrict__ WrI,
                   float* __restrict__ logits) {
    const int m = blockIdx.y;
    const float*  x  = m ? xi : xr;
    const float4* Wr = (const float4*)(m ? WrI : WrR);
    float a00 = 0.f, a01 = 0.f, a02 = 0.f, a03 = 0.f;
    float a10 = 0.f, a11 = 0.f, a12 = 0.f, a13 = 0.f;
    const int stride = gridDim.x * blockDim.x;
    for (int i = blockIdx.x * blockDim.x + threadIdx.x; i < NPIX; i += stride) {
        __builtin_prefetch(&Wr[i + 4 * stride], 0, 0);   // global_prefetch_b8
        float4 w = Wr[i];
        float x0 = x[i];
        float x1 = x[NPIX + i];
        a00 += x0 * w.x; a01 += x0 * w.y; a02 += x0 * w.z; a03 += x0 * w.w;
        a10 += x1 * w.x; a11 += x1 * w.y; a12 += x1 * w.z; a13 += x1 * w.w;
    }
    // wave32 shuffle reduction
    for (int off = 16; off > 0; off >>= 1) {
        a00 += __shfl_down(a00, off, 32); a01 += __shfl_down(a01, off, 32);
        a02 += __shfl_down(a02, off, 32); a03 += __shfl_down(a03, off, 32);
        a10 += __shfl_down(a10, off, 32); a11 += __shfl_down(a11, off, 32);
        a12 += __shfl_down(a12, off, 32); a13 += __shfl_down(a13, off, 32);
    }
    __shared__ float red[8][8];
    const int lane = threadIdx.x & 31, wv = threadIdx.x >> 5;
    if (lane == 0) {
        red[wv][0] = a00; red[wv][1] = a01; red[wv][2] = a02; red[wv][3] = a03;
        red[wv][4] = a10; red[wv][5] = a11; red[wv][6] = a12; red[wv][7] = a13;
    }
    __syncthreads();
    if (threadIdx.x < 8) {
        float s = 0.f;
        #pragma unroll
        for (int w = 0; w < 8; ++w) s += red[w][threadIdx.x];
        atomicAdd(&logits[m * 8 + threadIdx.x], s);   // [m][b][e], b*4+e = tid
    }
}

// ------------------------- Phase 2: gate -----------------------------------
__global__ void gate_kernel(float* wsf) {
    if (threadIdx.x != 0) return;
    const float* lg = wsf;
    float* wmod = wsf + 16;
    int*   sel  = (int*)(wsf + 20);
    float tops[2][2];
    for (int m = 0; m < 2; ++m)
        for (int b = 0; b < 2; ++b) {
            float v[4], p[4];
            float mx = -1e30f;
            for (int e = 0; e < 4; ++e) { v[e] = lg[m * 8 + b * 4 + e]; mx = fmaxf(mx, v[e]); }
            float s = 0.f;
            for (int e = 0; e < 4; ++e) { p[e] = __expf(v[e] - mx); s += p[e]; }
            for (int e = 0; e < 4; ++e) p[e] /= s;
            int i0 = 0;
            for (int e = 1; e < 4; ++e) if (p[e] > p[i0]) i0 = e;
            int i1 = (i0 == 0) ? 1 : 0;
            for (int e = 0; e < 4; ++e) if (e != i0 && p[e] > p[i1]) i1 = e;
            sel[(m * 2 + b) * 2 + 0] = i0;
            sel[(m * 2 + b) * 2 + 1] = i1;
            tops[m][b] = p[i0] + p[i1];
        }
    for (int b = 0; b < 2; ++b) {
        float s0 = tops[0][b], s1 = tops[1][b];
        float mx = fmaxf(s0, s1);
        float e0 = __expf(s0 - mx), e1 = __expf(s1 - mx);
        wmod[0 * 2 + b] = e0 / (e0 + e1);
        wmod[1 * 2 + b] = e1 / (e0 + e1);
    }
}

// ------------------------- Phase 3: fused experts --------------------------
// grid = (W/16, H/16, B); block = 256 threads = 8 wave32.
// Per block: accumulate all 8 expert evaluations (2 modalities x 2 experts x
// 2 inner-batch images) for one 16x16 spatial tile of out[b].
__global__ __launch_bounds__(256)
void mole_expert_kernel(const float* __restrict__ xrgb, const float* __restrict__ xir,
                        const float* __restrict__ w1r, const float* __restrict__ b1r,
                        const float* __restrict__ w2r, const float* __restrict__ b2r,
                        const float* __restrict__ w1i, const float* __restrict__ b1i,
                        const float* __restrict__ w2i, const float* __restrict__ b2i,
                        const int* __restrict__ sel, const float* __restrict__ wmodv,
                        float* __restrict__ out) {
    // LDS: 25600 + 10368 + 9216 + 9216 + 64 + 128 = 54592 B
    __shared__ __align__(16) _Float16 xs[20][20][32];   // input tile, halo 2, ch-fastest
    __shared__ __align__(16) _Float16 hs[18][18][16];   // hidden tile, halo 1
    __shared__ __align__(16) _Float16 w1s[16][9][32];   // [o][tap][cin]
    __shared__ __align__(16) _Float16 w2s[32][9][16];   // [o][tap][cin]
    __shared__ float b1s[16];
    __shared__ float b2s[32];

    const int tileC0 = blockIdx.x * 16;
    const int tileR0 = blockIdx.y * 16;
    const int bz     = blockIdx.z;
    const int tid    = threadIdx.x;
    const int lane   = tid & 31;
    const int wv     = tid >> 5;       // wave id 0..7
    const int hb     = lane >> 4;      // lane half (K-half select)
    const int ln     = lane & 15;      // M row / N column

    float outAcc[4][8];
    #pragma unroll
    for (int j = 0; j < 4; ++j)
        #pragma unroll
        for (int v = 0; v < 8; ++v) outAcc[j][v] = 0.f;

    for (int m = 0; m < 2; ++m) {
        const float* xsrc = m ? xir : xrgb;
        const float* w1g  = m ? w1i : w1r;
        const float* b1g  = m ? b1i : b1r;
        const float* w2g  = m ? w2i : w2r;
        const float* b2g  = m ? b2i : b2r;
        const float  wm   = wmodv[m * 2 + bz];

        for (int k = 0; k < 2; ++k) {
            const int e = sel[(m * 2 + bz) * 2 + k];
            __syncthreads();   // previous eval done with w*/hs/xs
            // ---- load expert weights (f32 -> f16, tap-major cin-fastest) ----
            {
                const float* w1p = w1g + (size_t)(bz * 4 + e) * (16 * 32 * 9);
                const float* w2p = w2g + (size_t)(bz * 4 + e) * (32 * 16 * 9);
                for (int f = tid; f < 16 * 9 * 32; f += 256) {
                    int o = f / 288, rem = f % 288, t = rem / 32, ci = rem % 32;
                    w1s[o][t][ci] = (_Float16)w1p[(o * 32 + ci) * 9 + t];
                }
                for (int f = tid; f < 32 * 9 * 16; f += 256) {
                    int o = f / 144, rem = f % 144, t = rem / 16, ci = rem % 16;
                    w2s[o][t][ci] = (_Float16)w2p[(o * 16 + ci) * 9 + t];
                }
                if (tid < 16)      b1s[tid]      = b1g[(bz * 4 + e) * 16 + tid];
                else if (tid < 48) b2s[tid - 16] = b2g[(bz * 4 + e) * 32 + (tid - 16)];
            }
            for (int bp = 0; bp < 2; ++bp) {
                if (bp) __syncthreads();
                // ---- load reflect-padded input tile (f32 -> f16) ----
                {
                    const float* xp = xsrc + (size_t)bp * C_IN * IMG_H * IMG_W;
                    for (int f = tid; f < 20 * 20 * 32; f += 256) {
                        int ch = f / 400, rem = f % 400, i = rem / 20, j = rem % 20;
                        int rr = reflectI(tileR0 + i - 2, IMG_H);
                        int cc = reflectI(tileC0 + j - 2, IMG_W);
                        xs[i][j][ch] = (_Float16)xp[((size_t)ch * IMG_H + rr) * IMG_W + cc];
                    }
                }
                __syncthreads();
                // ---- conv1: 18x18x16 hidden tile, K = 9 taps x 32 cin ----
                // 324 positions -> 21 N-groups of 16 (last partially dummy)
                for (int g = wv; g < 21; g += 8) {
                    v8f acc = zero8f();
                    int p = g * 16 + ln;
                    int r = p / 18, cq = p % 18;
                    const bool valid = (p < 324);
                    if (!valid) { r = 0; cq = 0; }
                    #pragma unroll
                    for (int t = 0; t < 9; ++t) {
                        const int dr = t / 3, dc = t % 3;
                        F16x16 a, bm;
                        // A 16x32 f16: lane<16 K=0..15, lane>=16 K=16..31 (contiguous)
                        const _Float16* wrow = &w1s[ln][t][0];
                        a.h8[0] = *(const v8h*)(wrow + hb * 16);
                        a.h8[1] = *(const v8h*)(wrow + hb * 16 + 8);
                        // B 32x16 f16: lane<16 K=0..15, lane>=16 K=16..31 (contiguous)
                        const _Float16* xpl = &xs[r + dr][cq + dc][hb * 16];
                        bm.h8[0] = *(const v8h*)xpl;
                        bm.h8[1] = *(const v8h*)(xpl + 8);
                        acc = __builtin_amdgcn_wmma_f32_16x16x32_f16(
                                  false, a.v, false, bm.v, (short)0, acc, false, false);
                    }
                    if (valid) {
                        F16x8 st;
                        #pragma unroll
                        for (int v = 0; v < 8; ++v) {   // D: M = v + 8*hb, N = ln
                            float hv = acc[v] + b1s[hb * 8 + v];
                            st.h[v] = (_Float16)fmaxf(hv, 0.f);
                        }
                        *(v8h*)&hs[r][cq][hb * 8] = st.v;
                    }
                }
                __syncthreads();
                // ---- conv2 + sigmoid accumulate: M=32 (2 tiles), K=144 (5 chunks) ----
                #pragma unroll
                for (int j = 0; j < 4; ++j) {
                    const int g  = wv + (j & 1) * 8;   // out row of tile
                    const int mt = j >> 1;             // M tile (out ch 0..15 / 16..31)
                    v8f acc = zero8f();
                    #pragma unroll
                    for (int c = 0; c < 5; ++c) {
                        const int t0 = 2 * c, t1 = t0 + 1;   // two taps per K chunk
                        F16x16 a, bm;
                        const int tA = hb ? t1 : t0;
                        if (tA < 9) { a.h8[0] = *(const v8h*)&w2s[mt * 16 + ln][tA][0];
                                      a.h8[1] = *(const v8h*)&w2s[mt * 16 + ln][tA][8]; }
                        else        { a.h8[0] = zero8h(); a.h8[1] = zero8h(); }
                        int tB  = hb ? t1 : t0;
                        int tBc = (tB < 9) ? tB : 0;
                        int dr = tBc / 3, dc = tBc % 3;
                        // reflect h at the image border (reference reflect-pads h)
                        int hr = reflectI(tileR0 + g  + dr - 1, IMG_H) - tileR0 + 1;
                        int hc = reflectI(tileC0 + ln + dc - 1, IMG_W) - tileC0 + 1;
                        const _Float16* hp = &hs[hr][hc][0];
                        bm.h8[0] = *(const v8h*)hp;
                        bm.h8[1] = *(const v8h*)(hp + 8);
                        if (tB >= 9) { bm.h8[0] = zero8h(); bm.h8[1] = zero8h(); }
                        acc = __builtin_amdgcn_wmma_f32_16x16x32_f16(
                                  false, a.v, false, bm.v, (short)0, acc, false, false);
                    }
                    #pragma unroll
                    for (int v = 0; v < 8; ++v) {
                        const int o = mt * 16 + hb * 8 + v;
                        float s = acc[v] + b2s[o];
                        outAcc[j][v] += wm * (1.0f / (1.0f + __expf(-s)));
                    }
                }
            } // bp
        } // k
    } // m
    // ---- final store: each block owns its tile exclusively ----
    #pragma unroll
    for (int j = 0; j < 4; ++j) {
        const int g   = wv + (j & 1) * 8;
        const int mt  = j >> 1;
        const int row = tileR0 + g;
        const int col = tileC0 + ln;
        #pragma unroll
        for (int v = 0; v < 8; ++v) {
            const int o = mt * 16 + hb * 8 + v;
            out[((size_t)(bz * C_IN + o) * IMG_H + row) * IMG_W + col] = outAcc[j][v];
        }
    }
}

// ---------------------------------------------------------------------------
extern "C" void kernel_launch(void* const* d_in, const int* in_sizes, int n_in,
                              void* d_out, int out_size, void* d_ws, size_t ws_size,
                              hipStream_t stream) {
    (void)in_sizes; (void)n_in; (void)out_size; (void)ws_size;
    const float* rgb = (const float*)d_in[0];
    const float* ir  = (const float*)d_in[1];
    const float* WrR = (const float*)d_in[2];
    const float* WrI = (const float*)d_in[3];
    const float* w1r = (const float*)d_in[4];
    const float* b1r = (const float*)d_in[5];
    const float* w2r = (const float*)d_in[6];
    const float* b2r = (const float*)d_in[7];
    const float* w1i = (const float*)d_in[8];
    const float* b1i = (const float*)d_in[9];
    const float* w2i = (const float*)d_in[10];
    const float* b2i = (const float*)d_in[11];
    float* wsf = (float*)d_ws;
    float* out = (float*)d_out;

    zero_kernel<<<1, 64, 0, stream>>>(wsf);
    router_kernel<<<dim3(1024, 2), 256, 0, stream>>>(rgb, ir, WrR, WrI, wsf);
    gate_kernel<<<1, 32, 0, stream>>>(wsf);
    const int*   sel  = (const int*)(wsf + 20);
    const float* wmod = wsf + 16;
    mole_expert_kernel<<<dim3(IMG_W / 16, IMG_H / 16, 2), 256, 0, stream>>>(
        rgb, ir, w1r, b1r, w2r, b2r, w1i, b1i, w2i, b2i, sel, wmod, out);
}
